// GPNN_27659589386647
// MI455X (gfx1250) — matrix operand order repaired and checked by reading, not compiled, not run on hardware
//
#include <hip/hip_runtime.h>

// ---------------------------------------------------------------------------
// GPNN forward for MI455X (gfx1250, wave32, WMMA bf16 -> f32 accumulate)
// All dense GEMMs and batched adjacency aggregations run through
// v_wmma_f32_16x16x32_bf16 with 2x2 register blocking per wave
// (32x32 wave tile, 4 accumulators, 2 b128 loads per WMMA).
// Workspace arena ~182 MB (harness d_ws).
// ---------------------------------------------------------------------------

#define EPS_F 1.1920928955078125e-07f

typedef __attribute__((ext_vector_type(16))) __bf16 v16bf;
typedef __attribute__((ext_vector_type(8)))  float  v8f;

__device__ __forceinline__ unsigned short f2bf(float f) {
  unsigned u = __float_as_uint(f);
  u += 0x7FFFu + ((u >> 16) & 1u);   // round-to-nearest-even
  return (unsigned short)(u >> 16);
}

struct alignas(16) U4 { unsigned a, b, c, d; };
union FragU { v16bf v; unsigned u[8]; U4 q[2]; };

// ---------------------------------------------------------------------------
// Generic WMMA GEMM:  C(M,Nc) = act( A(M,K) @ W(Nc,K)^T + bias )
//  - A: bf16 row-major, row stride K (batched adjacency collapses to this too)
//  - W: bf16 row-major (Nc,K); optional per-batch stride (batch = row/NB)
//  - Outputs: optional f32 C, optional bf16 C (ldc/colOff), optional
//    transposed bf16 store Ct[(batch*Nc + col)*NB + rowInBatch]
// Block = 256 threads = 8 waves (2 in M x 4 in N), block tile 64x128,
// wave tile 32x32 (2x2 register blocking of 16x16 WMMA tiles).
// ---------------------------------------------------------------------------
__global__ __launch_bounds__(256) void k_gemm(
    const unsigned short* __restrict__ A,
    const unsigned short* __restrict__ W, long long wBatchStride,
    const float* __restrict__ bias,
    float* __restrict__ Cf, unsigned short* __restrict__ Cb,
    int ldc, int colOff,
    unsigned short* __restrict__ Ct,
    int M, int Nc, int K, int NB, int act) {
  const int wave = threadIdx.x >> 5;
  const int lane = threadIdx.x & 31;
  const int rl = lane & 15;         // row (A) / col (B) within 16x16 subtile
  const int hi = lane >> 4;         // half-wave selector
  const int tileRow = blockIdx.x * 64 + (wave & 1) * 32;
  const int tileCol = blockIdx.y * 128 + (wave >> 1) * 32;
  const int batch = tileRow / NB;   // 32-row tile never crosses NB boundary

  const unsigned* A32a = (const unsigned*)(A + (size_t)(tileRow + rl) * K);
  const unsigned* A32b = (const unsigned*)(A + (size_t)(tileRow + 16 + rl) * K);
  const unsigned short* Wb = W + (size_t)batch * wBatchStride;
  const unsigned* W32a = (const unsigned*)(Wb + (size_t)(tileCol + rl) * K);
  const unsigned* W32b = (const unsigned*)(Wb + (size_t)(tileCol + 16 + rl) * K);

  v8f acc00 = {0.f, 0.f, 0.f, 0.f, 0.f, 0.f, 0.f, 0.f};
  v8f acc01 = acc00, acc10 = acc00, acc11 = acc00;

  const int Kd = K >> 1;  // dwords per row
  for (int k2 = 0; k2 < Kd; k2 += 16) {  // K step of 32 elements
    FragU fa0, fa1, fb0, fb1;
    // A frags: lanes 0-15 rows, VGPR0-3 K={8hi..8hi+7}, VGPR4-7 K={16+8hi..+7}
    fa0.q[0] = *(const U4*)(A32a + k2 + 4 * hi);
    fa0.q[1] = *(const U4*)(A32a + k2 + 8 + 4 * hi);
    fa1.q[0] = *(const U4*)(A32b + k2 + 4 * hi);
    fa1.q[1] = *(const U4*)(A32b + k2 + 8 + 4 * hi);
    // B frags (W^T): lanes 0-15 cols, VGPR0-3 K={8hi..8hi+7}, VGPR4-7 K={16+8hi..+7}
    fb0.q[0] = *(const U4*)(W32a + k2 + 4 * hi);
    fb0.q[1] = *(const U4*)(W32a + k2 + 8 + 4 * hi);
    fb1.q[0] = *(const U4*)(W32b + k2 + 4 * hi);
    fb1.q[1] = *(const U4*)(W32b + k2 + 8 + 4 * hi);
    acc00 = __builtin_amdgcn_wmma_f32_16x16x32_bf16(
        false, fa0.v, false, fb0.v, (short)0, acc00, false, false);
    acc01 = __builtin_amdgcn_wmma_f32_16x16x32_bf16(
        false, fa0.v, false, fb1.v, (short)0, acc01, false, false);
    acc10 = __builtin_amdgcn_wmma_f32_16x16x32_bf16(
        false, fa1.v, false, fb0.v, (short)0, acc10, false, false);
    acc11 = __builtin_amdgcn_wmma_f32_16x16x32_bf16(
        false, fa1.v, false, fb1.v, (short)0, acc11, false, false);
  }

  const int c0 = tileCol + rl;        // col of subtile j=0
  const int c1 = tileCol + 16 + rl;   // col of subtile j=1
  const float bv0 = bias ? bias[c0] : 0.0f;
  const float bv1 = bias ? bias[c1] : 0.0f;

#pragma unroll
  for (int si = 0; si < 2; ++si) {
    const int rowBase = tileRow + 16 * si + 8 * hi;
#pragma unroll
    for (int j = 0; j < 8; ++j) {
      const int gRow = rowBase + j;
      float v0 = (si == 0 ? acc00[j] : acc10[j]) + bv0;
      float v1 = (si == 0 ? acc01[j] : acc11[j]) + bv1;
      if (act == 1) {
        v0 = v0 > 0.f ? v0 : 0.f;
        v1 = v1 > 0.f ? v1 : 0.f;
      }
      if (Cf) {
        Cf[(size_t)gRow * ldc + colOff + c0] = v0;
        Cf[(size_t)gRow * ldc + colOff + c1] = v1;
      }
      if (Cb) {
        Cb[(size_t)gRow * ldc + colOff + c0] = f2bf(v0);
        Cb[(size_t)gRow * ldc + colOff + c1] = f2bf(v1);
      }
      if (Ct) {
        const int tb = gRow / NB, tr = gRow - tb * NB;
        Ct[((size_t)tb * Nc + c0) * NB + tr] = f2bf(v0);
        Ct[((size_t)tb * Nc + c1) * NB + tr] = f2bf(v1);
      }
    }
  }
}

// ---------------------------------------------------------------------------
// Weight conversion fp32 -> bf16 (direct and last-two-dims transpose)
// ---------------------------------------------------------------------------
__global__ void k_cvt(const float* __restrict__ s, unsigned short* __restrict__ d, int n) {
  int i = blockIdx.x * 256 + threadIdx.x;
  if (i < n) d[i] = f2bf(s[i]);
}
// src (E,R,C) -> dst (E,C,R)
__global__ void k_cvtT(const float* __restrict__ s, unsigned short* __restrict__ d,
                       int E, int R, int C) {
  int i = blockIdx.x * 256 + threadIdx.x;
  int tot = E * R * C;
  if (i >= tot) return;
  int e = i / (R * C);
  int rem = i - e * R * C;
  int c = rem / R;
  int r = rem - c * R;
  d[i] = f2bf(s[(e * R + r) * C + c]);
}

// ---------------------------------------------------------------------------
// Adjacency normalization.
//  k_norm_L:  L(B,N,N,E1) -> Ln(E1,B,N,N) bf16, binarized, row / (deg+eps)
//  k_norm_p:  Lp(B,N,N)   -> bf16, row / (rowsum+eps)
// One block per row (256 threads == N).
// ---------------------------------------------------------------------------
__global__ void k_norm_L(const float* __restrict__ L, unsigned short* __restrict__ Ln) {
  const int blk = blockIdx.x;   // = (e*64 + b)*256 + n
  const int m = threadIdx.x;
  const int n = blk & 255;
  const int t = blk >> 8;
  const int b = t & 63;
  const int e = t >> 6;
  float v = L[(((size_t)(b * 256 + n)) * 256 + m) * 4 + e];
  float x = (v != 0.0f) ? 1.0f : 0.0f;
  __shared__ float sh[256];
  sh[m] = x;
  __syncthreads();
  for (int s = 128; s > 0; s >>= 1) {
    if (m < s) sh[m] += sh[m + s];
    __syncthreads();
  }
  float inv = 1.0f / (sh[0] + EPS_F);
  Ln[(size_t)blk * 256 + m] = f2bf(x * inv);
}

__global__ void k_norm_p(const float* __restrict__ Lp, unsigned short* __restrict__ Lo) {
  const int blk = blockIdx.x;
  const int m = threadIdx.x;
  float v = Lp[(size_t)blk * 256 + m];
  __shared__ float sh[256];
  sh[m] = v;
  __syncthreads();
  for (int s = 128; s > 0; s >>= 1) {
    if (m < s) sh[m] += sh[m + s];
    __syncthreads();
  }
  Lo[(size_t)blk * 256 + m] = f2bf(v / (sh[0] + EPS_F));
}

// Embedding gather -> bf16 rows (M,128)
__global__ void k_gather(const int* __restrict__ nf, const float* __restrict__ emb,
                         unsigned short* __restrict__ out) {
  int i = blockIdx.x * 256 + threadIdx.x;  // < M*128
  int m = i >> 7, d = i & 127;
  out[i] = f2bf(emb[nf[m] * 128 + d]);
}

// GRU elementwise: gi,gh are (M,3H) f32 with biases already fused by GEMM.
__global__ void k_gru(const float* __restrict__ gi, const float* __restrict__ gh,
                      float* __restrict__ hf, unsigned short* __restrict__ hb, int tot) {
  int i = blockIdx.x * 256 + threadIdx.x;
  if (i >= tot) return;
  int m = i >> 7, h = i & 127;
  const float* gim = gi + (size_t)m * 384;
  const float* ghm = gh + (size_t)m * 384;
  float r = 1.f / (1.f + expf(-(gim[h] + ghm[h])));
  float z = 1.f / (1.f + expf(-(gim[128 + h] + ghm[128 + h])));
  float nn = tanhf(gim[256 + h] + r * ghm[256 + h]);
  float o = (1.f - z) * nn + z * hf[i];
  hf[i] = o;
  hb[i] = f2bf(o);
}

// cat = [state, sc, sx] along H (bf16 copies)
__global__ void k_cat(const unsigned short* __restrict__ st,
                      const unsigned short* __restrict__ sc,
                      const unsigned short* __restrict__ sx,
                      unsigned short* __restrict__ cat) {
  int i = blockIdx.x * 256 + threadIdx.x;  // < M*384
  int m = i / 384, c = i - m * 384;
  unsigned short v;
  if (c < 128)      v = st[(size_t)m * 128 + c];
  else if (c < 256) v = sc[(size_t)m * 128 + c - 128];
  else              v = sx[(size_t)m * 128 + c - 256];
  cat[i] = v;
}

// out[b,p] = mean_n( sigmoid(state.attW + attb) * (state.outW[p] + outb[p]) )
__global__ void k_final(const float* __restrict__ st, const float* __restrict__ attW,
                        const float* __restrict__ attb, const float* __restrict__ outW,
                        const float* __restrict__ outb, float* __restrict__ out) {
  const int b = blockIdx.x, n = threadIdx.x;
  const float* row = st + ((size_t)(b * 256 + n)) * 128;
  float accA = attb[0];
  float accY[10];
#pragma unroll
  for (int p = 0; p < 10; ++p) accY[p] = outb[p];
  for (int d = 0; d < 128; ++d) {
    float s = row[d];
    accA += s * attW[d];
#pragma unroll
    for (int p = 0; p < 10; ++p) accY[p] += s * outW[p * 128 + d];
  }
  float a = 1.f / (1.f + expf(-accA));
  __shared__ float sh[256];
  for (int p = 0; p < 10; ++p) {
    sh[n] = a * accY[p];
    __syncthreads();
    for (int s = 128; s > 0; s >>= 1) {
      if (n < s) sh[n] += sh[n + s];
      __syncthreads();
    }
    if (n == 0) out[b * 10 + p] = sh[0] / 256.0f;
    __syncthreads();
  }
}

// ---------------------------------------------------------------------------
// Host orchestration
// ---------------------------------------------------------------------------
extern "C" void kernel_launch(void* const* d_in, const int* in_sizes, int n_in,
                              void* d_out, int out_size, void* d_ws, size_t ws_size,
                              hipStream_t stream) {
  (void)in_sizes; (void)n_in; (void)out_size; (void)ws_size;
  constexpr int kB = 64, kN = 256, kH = 128, kE = 4, kM = kB * kN;  // 16384

  // ---- inputs (setup_inputs order) ----
  const int*   node_feat = (const int*)  d_in[0];
  const float* L         = (const float*)d_in[1];
  const float* L_cluster = (const float*)d_in[2];
  const float* L_cut     = (const float*)d_in[3];
  const float* emb       = (const float*)d_in[4];
  const float* W_in      = (const float*)d_in[5];
  const float* b_in      = (const float*)d_in[6];
  const float* msg_W1    = (const float*)d_in[7];
  const float* msg_b1    = (const float*)d_in[8];
  const float* msg_W2    = (const float*)d_in[9];
  const float* msg_b2    = (const float*)d_in[10];
  const float* gru_Wih   = (const float*)d_in[11];
  const float* gru_Whh   = (const float*)d_in[12];
  const float* gru_bih   = (const float*)d_in[13];
  const float* gru_bhh   = (const float*)d_in[14];
  const float* grup_Wih  = (const float*)d_in[15];
  const float* grup_Whh  = (const float*)d_in[16];
  const float* grup_bih  = (const float*)d_in[17];
  const float* grup_bhh  = (const float*)d_in[18];
  const float* st_W1     = (const float*)d_in[19];
  const float* st_b1     = (const float*)d_in[20];
  const float* st_W2     = (const float*)d_in[21];
  const float* st_b2     = (const float*)d_in[22];
  const float* att_W     = (const float*)d_in[23];
  const float* att_b     = (const float*)d_in[24];
  const float* out_W     = (const float*)d_in[25];
  const float* out_b     = (const float*)d_in[26];

  // ---- workspace arena (~182 MB) ----
  char* ws = (char*)d_ws;
  size_t off = 0;
  auto alloc = [&](size_t bytes) -> void* {
    void* p = ws + off;
    off = (off + bytes + 255) & ~(size_t)255;
    return p;
  };
  typedef unsigned short us;
  us* LnB  = (us*)alloc((size_t)kE * kB * kN * kN * 2);  // 33.5 MB
  us* LcB  = (us*)alloc((size_t)kB * kN * kN * 2);       //  8.4 MB
  us* LxB  = (us*)alloc((size_t)kB * kN * kN * 2);       //  8.4 MB
  us* WinB  = (us*)alloc(128 * 128 * 2);
  us* W1TB  = (us*)alloc(4 * 128 * 128 * 2);
  us* W2TB  = (us*)alloc(4 * 128 * 128 * 2);
  us* WihB  = (us*)alloc(384 * 512 * 2);
  us* WhhB  = (us*)alloc(384 * 128 * 2);
  us* pWihB = (us*)alloc(384 * 128 * 2);
  us* pWhhB = (us*)alloc(384 * 128 * 2);
  us* S1B   = (us*)alloc(512 * 384 * 2);
  us* S2B   = (us*)alloc(128 * 512 * 2);
  us*    stB = (us*)   alloc((size_t)kM * 128 * 2);
  float* stF = (float*)alloc((size_t)kM * 128 * 4);
  us*    scB = (us*)   alloc((size_t)kM * 128 * 2);
  float* scF = (float*)alloc((size_t)kM * 128 * 4);
  us*    sxB = (us*)   alloc((size_t)kM * 128 * 2);
  float* sxF = (float*)alloc((size_t)kM * 128 * 4);
  us* tmpB = (us*)alloc((size_t)kM * 128 * 2);
  us* aggB = (us*)alloc((size_t)kM * 128 * 2);
  us* mTB  = (us*)alloc((size_t)kE * kB * kH * kN * 2);  // 16.8 MB
  us* msgB = (us*)alloc((size_t)kM * 512 * 2);           // 16.8 MB
  float* giF = (float*)alloc((size_t)kM * 384 * 4);      // 25.2 MB
  float* ghF = (float*)alloc((size_t)kM * 384 * 4);      // 25.2 MB
  // alias (liveness-disjoint): cat over gi, hidden(512) over gh
  us* catB = (us*)giF;
  us* hidB = (us*)ghF;

  auto cvt = [&](const float* s, us* d, int n) {
    k_cvt<<<(n + 255) / 256, 256, 0, stream>>>(s, d, n);
  };
  auto gemm = [&](const us* A, const us* W, long long wbs, const float* bias,
                  float* Cf, us* Cb, int ldc, int colOff, us* Ct,
                  int Nc, int K, int NB, int act) {
    dim3 g(kM / 64, Nc / 128);
    k_gemm<<<g, 256, 0, stream>>>(A, W, wbs, bias, Cf, Cb, ldc, colOff, Ct,
                                  kM, Nc, K, NB, act);
  };
  auto gru = [&](const float* gi, const float* gh, float* hf, us* hb) {
    k_gru<<<(kM * 128) / 256, 256, 0, stream>>>(gi, gh, hf, hb, kM * 128);
  };

  // ---- one-time weight conversion ----
  cvt(W_in, WinB, 128 * 128);
  k_cvtT<<<(4 * 128 * 128 + 255) / 256, 256, 0, stream>>>(msg_W1, W1TB, 4, 128, 128);
  k_cvtT<<<(4 * 128 * 128 + 255) / 256, 256, 0, stream>>>(msg_W2, W2TB, 4, 128, 128);
  cvt(gru_Wih,  WihB,  384 * 512);
  cvt(gru_Whh,  WhhB,  384 * 128);
  cvt(grup_Wih, pWihB, 384 * 128);
  cvt(grup_Whh, pWhhB, 384 * 128);
  cvt(st_W1, S1B, 512 * 384);
  cvt(st_W2, S2B, 128 * 512);

  // ---- adjacency normalization ----
  k_norm_L<<<kE * kB * kN, 256, 0, stream>>>(L, LnB);
  k_norm_p<<<kB * kN, 256, 0, stream>>>(L_cluster, LcB);
  k_norm_p<<<kB * kN, 256, 0, stream>>>(L_cut, LxB);

  // ---- initial state: emb[node_feat] @ W_in^T + b_in ----
  k_gather<<<(kM * 128) / 256, 256, 0, stream>>>(node_feat, emb, tmpB);
  gemm(tmpB, WinB, 0, b_in, stF, stB, 128, 0, nullptr, 128, 128, kM, 0);

  // one partition-propagation step: s <- GRU(Lmat @ MLP0(s), s; grup)
  auto part = [&](us* sB, float* sF, const us* Lmat) {
    gemm(sB, W1TB, 0, msg_b1, nullptr, tmpB, 128, 0, nullptr, 128, 128, kM, 1);
    gemm(tmpB, W2TB, 0, msg_b2, nullptr, nullptr, 0, 0, mTB, 128, 128, kN, 0);
    gemm(Lmat, mTB, (long long)kH * kN, nullptr, nullptr, aggB, 128, 0, nullptr,
         128, kN, kN, 0);
    gemm(aggB, pWihB, 0, grup_bih, giF, nullptr, 384, 0, nullptr, 384, 128, kM, 0);
    gemm(sB, pWhhB, 0, grup_bhh, ghF, nullptr, 384, 0, nullptr, 384, 128, kM, 0);
    gru(giF, ghF, sF, sB);
  };

  for (int it = 0; it < 3; ++it) {
    // sc = state ; sx = state
    hipMemcpyAsync(scF, stF, (size_t)kM * 128 * 4, hipMemcpyDeviceToDevice, stream);
    hipMemcpyAsync(scB, stB, (size_t)kM * 128 * 2, hipMemcpyDeviceToDevice, stream);
    hipMemcpyAsync(sxF, stF, (size_t)kM * 128 * 4, hipMemcpyDeviceToDevice, stream);
    hipMemcpyAsync(sxB, stB, (size_t)kM * 128 * 2, hipMemcpyDeviceToDevice, stream);
    part(scB, scF, LcB);
    part(scB, scF, LcB);
    part(sxB, sxF, LxB);
    part(sxB, sxF, LxB);

    // state = relu(cat @ st_W1^T + b1) @ st_W2^T + b2
    k_cat<<<(kM * 384) / 256, 256, 0, stream>>>(stB, scB, sxB, catB);
    gemm(catB, S1B, 0, st_b1, nullptr, hidB, 512, 0, nullptr, 512, 384, kM, 1);
    gemm(hidB, S2B, 0, st_b2, stF, stB, 128, 0, nullptr, 128, 512, kM, 0);

    // full prop: per-edge-type MLP, Ln aggregate, concat, GRU(gru_*)
    for (int e = 0; e < kE; ++e) {
      gemm(stB, W1TB + (size_t)e * 128 * 128, 0, msg_b1 + e * 128,
           nullptr, tmpB, 128, 0, nullptr, 128, 128, kM, 1);
      gemm(tmpB, W2TB + (size_t)e * 128 * 128, 0, msg_b2 + e * 128,
           nullptr, nullptr, 0, 0, mTB + (size_t)e * kB * kH * kN,
           128, 128, kN, 0);
      gemm(LnB + (size_t)e * kB * kN * kN, mTB + (size_t)e * kB * kH * kN,
           (long long)kH * kN, nullptr, nullptr, msgB, 512, e * 128, nullptr,
           128, kN, kN, 0);
    }
    gemm(msgB, WihB, 0, gru_bih, giF, nullptr, 384, 0, nullptr, 384, 512, kM, 0);
    gemm(stB, WhhB, 0, gru_bhh, ghF, nullptr, 384, 0, nullptr, 384, 128, kM, 0);
    gru(giF, ghF, stF, stB);
  }

  // ---- readout ----
  k_final<<<kB, 256, 0, stream>>>(stF, att_W, att_b, out_W, out_b, (float*)d_out);
}
